// GAL_18004502905251
// MI455X (gfx1250) — matrix operand order repaired and checked by reading, not compiled, not run on hardware
//
#include <hip/hip_runtime.h>

typedef _Float16 h16;
typedef __attribute__((ext_vector_type(16))) _Float16 v16h;
typedef __attribute__((ext_vector_type(8)))  _Float16 v8h;
typedef __attribute__((ext_vector_type(2)))  _Float16 h2;
typedef __attribute__((ext_vector_type(8)))  float    v8f;
typedef __attribute__((ext_vector_type(4)))  float    v4f;

#define NROW 8192
#define DIM  256
#define UDIM 128
#define MINN 1e-15f
// fixed softmax shift (softmax is shift-invariant; dist bounded in [0,16.81]):
//   p = exp(dist - 8) = e^-8 * (1+mn)/(1-mn)   -- no log/exp needed
#define EM8   0.000335462628f    // e^-8
#define BIGR  1.9999998e7f       // (2-1e-7)/1e-7 : ratio at the artanh clip point

union AFrag { v16h v; h2 p[8]; };

__device__ __forceinline__ v8f wmma16(v16h a, v16h b, v8f c) {
  return __builtin_amdgcn_wmma_f32_16x16x32_f16(false, a, false, b, (short)0, c,
                                                false, false);
}

// xor-swizzle add/reduce across the 16-lane half that holds one C-tile row
template <int MASK>
__device__ __forceinline__ float swz_add(float v) {
  return v + __int_as_float(
      __builtin_amdgcn_ds_swizzle(__float_as_int(v), 0x1f | (MASK << 10)));
}
__device__ __forceinline__ float redsum16(float v) {
  v = swz_add<8>(v); v = swz_add<4>(v); v = swz_add<2>(v); v = swz_add<1>(v);
  return v;
}

// A-matrix 16x32 f16 layout (contiguous k = 16*kh + s): element-pair base K for VGPR vg, lane-half kh
__device__ __forceinline__ int apat(int vg, int kh) {
  return kh * 16 + vg * 2;
}

// softmax weight p = exp(dist - 8) as a rational function of the Gram value:
//   mn = sqrt(num)/den (mobius norm), dist = 2*artanh(mn)
//   exp(dist-8) = e^-8 * (den + sqrt(num)) / (den - sqrt(num))
// masked (adj==0) or dist==0 (mn==0, incl. diagonal) -> 0, per reference.
__device__ __forceinline__ float hyp_p(float xy, float x2i, float x2j, int av) {
  float A = 1.f - 2.f * xy + x2j;
  float B = 1.f - x2i;
  float num = A * A * x2i - 2.f * A * B * xy + B * B * x2j;
  float den = fmaxf(1.f - 2.f * xy + x2i * x2j, MINN);
  float s = __builtin_sqrtf(fmaxf(num, 0.f));
  float d = den - s;
  float ratio = (d > 0.f) ? (den + s) / d : BIGR;
  ratio = fminf(ratio, BIGR);                    // == artanh clip at 1-1e-7
  return (av != 0 && s > 0.f) ? EM8 * ratio : 0.f;
}

// ---------------- bias -> expmap0(bias) ----------------
__global__ void bias_expmap_kernel(const float* __restrict__ bias,
                                   float* __restrict__ bvec,
                                   float* __restrict__ b2) {
  __shared__ float red[128];
  const int t = threadIdx.x;
  float v = bias[t];
  red[t] = v * v;
  __syncthreads();
  for (int s = 64; s > 0; s >>= 1) {
    if (t < s) red[t] += red[t + s];
    __syncthreads();
  }
  float nn = fmaxf(__builtin_sqrtf(red[0]), MINN);
  float e2 = __expf(2.f * nn);
  float th = 1.f - 2.f / (e2 + 1.f);        // tanh(nn)
  bvec[t] = (th / nn) * v;
  if (t == 0) *b2 = th * th;
}

// ---------------- x = expmap0(features @ kernel) ----------------
__global__ __launch_bounds__(128) void proj_expmap_kernel(
    const float* __restrict__ feat, const float* __restrict__ ker,
    h16* __restrict__ xh, float* __restrict__ x2) {
  __shared__ h16 skBt[UDIM][32];   // kernel slab, transposed, f16 (8KB)
  const int tid  = threadIdx.x;
  const int lane = tid & 31;
  const int nl = lane & 15, kh = lane >> 4, mb = kh * 8;
  const int i0 = blockIdx.x * 64 + (tid >> 5) * 16;

  const v8f vz = {0.f, 0.f, 0.f, 0.f, 0.f, 0.f, 0.f, 0.f};
  v8f acc[8];
#pragma unroll
  for (int t = 0; t < 8; ++t) acc[t] = vz;

  for (int k0 = 0; k0 < DIM; k0 += 32) {
    __syncthreads();
    {  // stage ker[k0..k0+32)[0..128) transposed into LDS as f16
      const int kk = tid >> 2;
      const int u0 = (tid & 3) * 32;
      const float* src = ker + (size_t)(k0 + kk) * UDIM + u0;
#pragma unroll
      for (int q = 0; q < 8; ++q) {
        v4f f = *(const v4f*)(src + q * 4);
#pragma unroll
        for (int e = 0; e < 4; ++e) skBt[u0 + q * 4 + e][kk] = (h16)f[e];
      }
    }
    __syncthreads();

    AFrag af;
    const float* frow = feat + (size_t)(i0 + nl) * DIM + k0;
#pragma unroll
    for (int vg = 0; vg < 8; ++vg) {
      int k = apat(vg, kh);
      h2 t = {(h16)frow[k], (h16)frow[k + 1]};
      af.p[vg] = t;
    }
#pragma unroll
    for (int ut = 0; ut < 8; ++ut) {
      v16h b = *(const v16h*)&skBt[ut * 16 + nl][kh * 16];
      acc[ut] = wmma16(af.v, b, acc[ut]);
    }
  }

#pragma unroll
  for (int r = 0; r < 8; ++r) {
    float n2 = 0.f;
#pragma unroll
    for (int t = 0; t < 8; ++t) { float h = acc[t][r]; n2 += h * h; }
    n2 = redsum16(n2);
    float nn = fmaxf(__builtin_sqrtf(n2), MINN);
    float e2 = __expf(2.f * nn);
    float th = 1.f - 2.f / (e2 + 1.f);    // tanh(nn)
    float sc = th / nn;
    const int row = i0 + mb + r;
    h16* orow = xh + (size_t)row * UDIM;
#pragma unroll
    for (int t = 0; t < 8; ++t) orow[t * 16 + nl] = (h16)(sc * acc[t][r]);
    if (nl == 0) x2[row] = th * th;
  }
}

// ---------------- fused hyperbolic attention ----------------
// 8 waves/WG = 4 query strips x 2 key-range splits; fixed-shift softmax with
// rational weights (no transcendentals in main loop); row sums via an extra
// WMMA against an all-ones B matrix; single pass over adj (HBM floor).
__global__ __launch_bounds__(256) void hyp_attn_kernel(
    const int* __restrict__ adj, const h16* __restrict__ xh,
    const float* __restrict__ x2, const float* __restrict__ bvec,
    const float* __restrict__ b2p, float* __restrict__ out) {
  // manual LDS layout (41.5KB); K/V tile region is overlaid by the combine bufs
  __shared__ __align__(32) char smem[41472];
  h16*   sV    = (h16*)smem;               // [2][32][128]  (16KB)
  h16*   sVt   = (h16*)(smem + 16384);     // [2][128][32]  (16KB)
  h16*   sP    = (h16*)(smem + 32768);     // [8][16][32]   (8KB)
  float* sX2j  = (float*)(smem + 40960);   // [2][32]
  float* sLred = (float*)(smem + 41216);   // [64]
  float* sOred = (float*)smem;             // overlay: [4][16][128] f32 (32KB)

  const int tid  = threadIdx.x;
  const int lane = tid & 31;
  const int wave = tid >> 5;
  const int sg   = wave >> 2;       // key-range split group (0/1)
  const int wq   = wave & 3;        // query strip within block
  const int gtid = tid & 127;       // thread id within split group
  const int nl = lane & 15, kh = lane >> 4, mb = kh * 8;
  const int i0 = blockIdx.x * 64 + wq * 16;
  const int jbase = sg * (NROW / 2);

  // resident Q fragments (A layout): 16 rows x 128 f16
  AFrag qf[4];
  {
    const h16* qrow = xh + (size_t)(i0 + nl) * UDIM;
#pragma unroll
    for (int uc = 0; uc < 4; ++uc)
#pragma unroll
      for (int vg = 0; vg < 8; ++vg)
        qf[uc].p[vg] = *(const h2*)(qrow + uc * 32 + apat(vg, kh));
  }
  float x2i[8];
#pragma unroll
  for (int r = 0; r < 8; ++r) x2i[r] = x2[i0 + mb + r];

  // all-ones B fragment: C[m][n] = sum_k P[m][k], replicated over lanes
  v16h vones;
#pragma unroll
  for (int e = 0; e < 16; ++e) vones[e] = (h16)1.0f;

  const v8f vz = {0.f, 0.f, 0.f, 0.f, 0.f, 0.f, 0.f, 0.f};
  v8f O[8], Lacc = vz;
#pragma unroll
  for (int t = 0; t < 8; ++t) O[t] = vz;

  for (int it = 0; it < (NROW / 2) / 32; ++it) {
    const int j0 = jbase + it * 32;
    __syncthreads();
    {  // each split group stages its own 32x128 f16 key tile (both layouts)
      const int jr = gtid >> 2;
      const int u0 = (gtid & 3) * 32;
      const h16* src = xh + (size_t)(j0 + jr) * UDIM + u0;
      h16* dV  = sV  + sg * 4096 + jr * 128;
      h16* dVt = sVt + sg * 4096 + jr;
#pragma unroll
      for (int q = 0; q < 4; ++q) {
        v8h v = *(const v8h*)(src + q * 8);
        *(v8h*)(dV + u0 + q * 8) = v;
#pragma unroll
        for (int e = 0; e < 8; ++e) dVt[(u0 + q * 8 + e) * 32] = v[e];
      }
      if (gtid < 32) sX2j[sg * 32 + gtid] = x2[j0 + gtid];
    }
    __syncthreads();

    // S = Q . K^T : two 16x16 f32 tiles (K-dim = U = 128)
    v8f s0 = vz, s1 = vz;
    const h16* bV = sV + sg * 4096;
#pragma unroll
    for (int uc = 0; uc < 4; ++uc) {
      v16h b0 = *(const v16h*)(bV + nl * 128 + uc * 32 + kh * 16);
      v16h b1 = *(const v16h*)(bV + (16 + nl) * 128 + uc * 32 + kh * 16);
      s0 = wmma16(qf[uc].v, b0, s0);
      s1 = wmma16(qf[uc].v, b1, s1);
    }

    // rational softmax weights + adjacency mask -> P (f16) in LDS
    const float x2j0 = sX2j[sg * 32 + nl], x2j1 = sX2j[sg * 32 + 16 + nl];
    h16* pP = sP + wave * 512;
#pragma unroll
    for (int r = 0; r < 8; ++r) {
      const size_t arow = (size_t)(i0 + mb + r) * NROW + j0;
      int av0 = adj[arow + nl];
      int av1 = adj[arow + 16 + nl];
      if (it + 1 < (NROW / 2) / 32) __builtin_prefetch(adj + arow + 32 + nl, 0, 0);
      pP[(mb + r) * 32 + nl]      = (h16)hyp_p(s0[r], x2i[r], x2j0, av0);
      pP[(mb + r) * 32 + 16 + nl] = (h16)hyp_p(s1[r], x2i[r], x2j1, av1);
    }
    asm volatile("s_wait_dscnt 0" ::: "memory");  // same-wave LDS RAW + alias fence

    // O += P.V ; Lacc += P.ones (row sums)
    AFrag pf;
#pragma unroll
    for (int vg = 0; vg < 8; ++vg)
      pf.p[vg] = *(const h2*)(pP + nl * 32 + apat(vg, kh));
    const h16* bVt = sVt + sg * 4096;
#pragma unroll
    for (int t = 0; t < 8; ++t) {
      v16h vb = *(const v16h*)(bVt + (t * 16 + nl) * 32 + kh * 16);
      O[t] = wmma16(pf.v, vb, O[t]);
    }
    Lacc = wmma16(pf.v, vones, Lacc);
  }

  // combine the two key-range splits through LDS (overlaying the K/V tiles)
  __syncthreads();
  if (sg == 1) {
    float* dst = sOred + (wq * 16 + mb) * UDIM;
#pragma unroll
    for (int r = 0; r < 8; ++r) {
#pragma unroll
      for (int t = 0; t < 8; ++t) dst[r * UDIM + t * 16 + nl] = O[t][r];
      if (nl == 0) sLred[wq * 16 + mb + r] = Lacc[r];
    }
  }
  __syncthreads();
  if (sg != 0) return;

  const float* srcO = sOred + (wq * 16 + mb) * UDIM;
  float lsum[8];
#pragma unroll
  for (int r = 0; r < 8; ++r) {
#pragma unroll
    for (int t = 0; t < 8; ++t) O[t][r] = O[t][r] + srcO[r * UDIM + t * 16 + nl];
    lsum[r] = Lacc[r] + sLred[wq * 16 + mb + r];
  }

  // epilogue: normalize, mobius matvec rescale, mobius bias add
  const float b2 = *b2p;
  float bv[8];
#pragma unroll
  for (int t = 0; t < 8; ++t) bv[t] = bvec[t * 16 + nl];

#pragma unroll
  for (int r = 0; r < 8; ++r) {
    float inv = 1.f / lsum[r];
    float mx[8], mn2 = 0.f;
#pragma unroll
    for (int t = 0; t < 8; ++t) { mx[t] = O[t][r] * inv; mn2 += mx[t] * mx[t]; }
    mn2 = redsum16(mn2);
    float mxn = fmaxf(__builtin_sqrtf(mn2), MINN);
    float xn  = fmaxf(__builtin_sqrtf(x2i[r]), MINN);
    float z = fminf(xn, 1.f - 1e-7f);
    float at = 0.5f * __logf((1.f + z) / (1.f - z));   // artanh(xn)
    float arg = mxn / xn * at;
    float e2 = __expf(2.f * arg);
    float th = 1.f - 2.f / (e2 + 1.f);                 // tanh(arg)
    float fac = th / mxn;
    float ov[8], ob = 0.f;
#pragma unroll
    for (int t = 0; t < 8; ++t) { ov[t] = fac * mx[t]; ob += ov[t] * bv[t]; }
    ob = redsum16(ob);
    float o2 = fac * fac * mn2;
    float c1 = 1.f + 2.f * ob + b2;
    float c2 = 1.f - o2;
    float den = fmaxf(1.f + 2.f * ob + o2 * b2, MINN);
    float* orow = out + (size_t)(i0 + mb + r) * UDIM;
#pragma unroll
    for (int t = 0; t < 8; ++t)
      orow[t * 16 + nl] = (c1 * ov[t] + c2 * bv[t]) / den;
  }
}

extern "C" void kernel_launch(void* const* d_in, const int* in_sizes, int n_in,
                              void* d_out, int out_size, void* d_ws, size_t ws_size,
                              hipStream_t stream) {
  (void)in_sizes; (void)n_in; (void)out_size; (void)ws_size;
  const float* feat = (const float*)d_in[0];   // [8192,256]
  const int*   adj  = (const int*)d_in[1];     // [8192,8192]
  const float* ker  = (const float*)d_in[2];   // [256,128]
  const float* bias = (const float*)d_in[3];   // [128]
  float* out = (float*)d_out;                  // [8192,128]

  char* ws = (char*)d_ws;                      // needs ~2.13 MB
  h16*   xh   = (h16*)ws;                      // 8192*128 f16 = 2 MB
  float* x2   = (float*)(ws + (size_t)NROW * UDIM * sizeof(h16));
  float* bvec = x2 + NROW;
  float* b2   = bvec + UDIM;

  bias_expmap_kernel<<<1, 128, 0, stream>>>(bias, bvec, b2);
  proj_expmap_kernel<<<NROW / 64, 128, 0, stream>>>(feat, ker, xh, x2);
  hyp_attn_kernel<<<NROW / 64, 256, 0, stream>>>(adj, xh, x2, bvec, b2, out);
}
